// Transformer_35545149342138
// MI455X (gfx1250) — matrix operand order repaired and checked by reading, not compiled, not run on hardware
//
#include <hip/hip_runtime.h>
#include <hip/hip_bf16.h>

// ---------------------------------------------------------------------------
// Problem constants (from reference): B=16, N=512, E=512, H=8, MLP=2048,
// DEPTH=4, D=64, C4=128
// ---------------------------------------------------------------------------
#define BB 16
#define NN_ 512
#define EE 512
#define HH 8
#define MLP_ 2048
#define DEPTH_ 4
#define DD 64
#define C4_ 128

typedef __attribute__((ext_vector_type(16))) __bf16 v16bf;
typedef __attribute__((ext_vector_type(8)))  float  v8f;
typedef __attribute__((ext_vector_type(4)))  int    i4_t;
typedef __attribute__((address_space(1))) i4_t* g_i4p;   // global int4*
typedef __attribute__((address_space(3))) i4_t* l_i4p;   // LDS int4*

union U32u { uint4 q[2]; v16bf v; unsigned short s[16]; };
union U64u { uint4 q[4]; unsigned short s[32]; };

#if __has_builtin(__builtin_amdgcn_global_load_async_to_lds_b128)
#define HAS_ASYNC 1
#else
#define HAS_ASYNC 0
#endif

// 16-byte global -> LDS copy: async (ASYNCcnt / no VGPR roundtrip) if available
__device__ __forceinline__ void cp16(const unsigned short* g, unsigned short* l) {
#if HAS_ASYNC
    __builtin_amdgcn_global_load_async_to_lds_b128(
        (g_i4p)(i4_t*)g, (l_i4p)(i4_t*)l, 0, 0);
#else
    *(uint4*)l = *(const uint4*)g;
#endif
}
__device__ __forceinline__ void async_wait() {
#if HAS_ASYNC
#if __has_builtin(__builtin_amdgcn_s_wait_asynccnt)
    __builtin_amdgcn_s_wait_asynccnt(0);
#else
    asm volatile("s_wait_asynccnt 0" ::: "memory");
#endif
#endif
}

__device__ __forceinline__ unsigned short f2bf(float f) {
    unsigned int u = __builtin_bit_cast(unsigned int, f);
    u = (u + 0x7FFFu + ((u >> 16) & 1u)) >> 16;
    return (unsigned short)u;
}
__device__ __forceinline__ float gelu_f(float x) {
    return 0.5f * x * (1.0f + erff(x * 0.70710678118654752f));
}
__device__ __forceinline__ float sigm_f(float x) {
    return 1.0f / (1.0f + __expf(-x));
}
__device__ __forceinline__ v8f wmma_bf16(v16bf a, v16bf b, v8f c) {
    return __builtin_amdgcn_wmma_f32_16x16x32_bf16(false, a, false, b,
                                                   (short)0, c, false, false);
}

// ---------------------------------------------------------------------------
// fp32 -> bf16 elementwise
// ---------------------------------------------------------------------------
__global__ void k_cvt(const float* __restrict__ s, unsigned short* __restrict__ d, int n) {
    int i = blockIdx.x * blockDim.x + threadIdx.x;
    if (i < n) d[i] = f2bf(s[i]);
}

// ---------------------------------------------------------------------------
// LayerNorm: one row (E=512) per block, 256 threads, bf16 output
// ---------------------------------------------------------------------------
__global__ __launch_bounds__(256)
void k_layernorm(const float* __restrict__ x, const float* __restrict__ g,
                 const float* __restrict__ b, unsigned short* __restrict__ outb) {
    int row = blockIdx.x;
    const float* xr = x + (size_t)row * EE;
    int t = threadIdx.x;
    float v0 = xr[t], v1 = xr[t + 256];
    __shared__ float red[256];
    red[t] = v0 + v1;
    __syncthreads();
    for (int st = 128; st > 0; st >>= 1) {
        if (t < st) red[t] += red[t + st];
        __syncthreads();
    }
    float mean = red[0] * (1.0f / EE);
    __syncthreads();
    float d0 = v0 - mean, d1 = v1 - mean;
    red[t] = d0 * d0 + d1 * d1;
    __syncthreads();
    for (int st = 128; st > 0; st >>= 1) {
        if (t < st) red[t] += red[t + st];
        __syncthreads();
    }
    float rstd = rsqrtf(red[0] * (1.0f / EE) + 1e-5f);
    unsigned short* o = outb + (size_t)row * EE;
    o[t]       = f2bf(d0 * rstd * g[t]       + b[t]);
    o[t + 256] = f2bf(d1 * rstd * g[t + 256] + b[t + 256]);
}

// ---------------------------------------------------------------------------
// Tiled bf16 WMMA GEMM, double-buffered LDS with async global->LDS staging.
//  C[M,N] = act( A[M,K] * B' + bias ) + resid
//  bmode==0 (NT): B logical [N,K] row-major (weights, K contiguous)
//  bmode==1 (NN): B logical [K,N] row-major (activations as K x N)
//  bias_mode: 0 none, 1 per-col (bias[n]), 2 per-row (bias[m])
//  act: 0 none, 1 gelu.  outf (fp32) and outb (bf16) each optional.
//  Block tile 64(M) x 128(N), K step 32. 256 threads = 8 waves, wave tile 32x32.
// ---------------------------------------------------------------------------
__device__ __forceinline__ void stage_tiles(const unsigned short* __restrict__ A, int lda, int m0,
                                            const unsigned short* __restrict__ B, int ldb, int n0,
                                            int k0, int K, int bmode, int tid,
                                            unsigned short* sA, unsigned short* sB) {
    {
        int row = tid >> 2, ch = (tid & 3) * 8;
        cp16(A + (size_t)(m0 + row) * lda + k0 + ch, sA + row * 32 + ch);
    }
    if (bmode == 0) {
        int n = tid >> 1, hh = (tid & 1) * 16;
        const unsigned short* src = B + (size_t)(n0 + n) * ldb + k0 + hh;
        cp16(src,     sB + n * 32 + hh);
        cp16(src + 8, sB + n * 32 + hh + 8);
    } else {
        int k = tid >> 3, nc = (tid & 7) * 16;
        const unsigned short* src = B + (size_t)(k0 + k) * ldb + n0 + nc;
        U32u tt;
        tt.q[0] = *(const uint4*)src;
        tt.q[1] = *(const uint4*)(src + 8);
        if (k0 + 32 < K) __builtin_prefetch(src + (size_t)32 * ldb, 0, 0);
#pragma unroll
        for (int e = 0; e < 16; ++e) sB[(nc + e) * 32 + k] = tt.s[e];
    }
}

__global__ __launch_bounds__(256)
void k_gemm(const unsigned short* __restrict__ A, int lda, long long batchA,
            const unsigned short* __restrict__ B, int ldb, long long batchB,
            const float* __restrict__ bias, int bias_mode,
            const float* __restrict__ resid,
            float* __restrict__ outf, unsigned short* __restrict__ outb,
            int ldc, long long batchC,
            int M, int N, int K, int bmode, int act) {
    int n0 = blockIdx.x * 128;
    int m0 = blockIdx.y * 64;
    long long zb = blockIdx.z;
    A += zb * batchA;
    B += zb * batchB;
    size_t cb = (size_t)(zb * batchC);

    __shared__ unsigned short sA[2][64 * 32];
    __shared__ unsigned short sB[2][128 * 32];

    int tid = threadIdx.x;
    int wave = tid >> 5, lane = tid & 31, half = lane >> 4, lm = lane & 15;
    int wrow = wave >> 2, wcol = wave & 3;

    v8f acc[2][2] = {};

    int buf = 0;
    stage_tiles(A, lda, m0, B, ldb, n0, 0, K, bmode, tid, sA[0], sB[0]);

    for (int k0 = 0; k0 < K; k0 += 32) {
        async_wait();
        __syncthreads();
        if (k0 + 32 < K)
            stage_tiles(A, lda, m0, B, ldb, n0, k0 + 32, K, bmode, tid,
                        sA[buf ^ 1], sB[buf ^ 1]);

        // ---- fragments from current buffer ----
        U32u af[2], bfrag[2];
#pragma unroll
        for (int mi = 0; mi < 2; ++mi) {
            int mrow = wrow * 32 + mi * 16 + lm;
            af[mi].q[0] = *(const uint4*)(sA[buf] + mrow * 32 + half * 8);
            af[mi].q[1] = *(const uint4*)(sA[buf] + mrow * 32 + half * 8 + 16);
        }
#pragma unroll
        for (int ni = 0; ni < 2; ++ni) {
            int n = wcol * 32 + ni * 16 + lm;
            bfrag[ni].q[0] = *(const uint4*)(sB[buf] + n * 32 + half * 8);
            bfrag[ni].q[1] = *(const uint4*)(sB[buf] + n * 32 + half * 8 + 16);
        }
#pragma unroll
        for (int mi = 0; mi < 2; ++mi)
#pragma unroll
            for (int ni = 0; ni < 2; ++ni)
                acc[mi][ni] = wmma_bf16(af[mi].v, bfrag[ni].v, acc[mi][ni]);
        buf ^= 1;
    }

    // ---- epilogue ----
#pragma unroll
    for (int mi = 0; mi < 2; ++mi)
#pragma unroll
        for (int ni = 0; ni < 2; ++ni) {
            int gn = n0 + wcol * 32 + ni * 16 + lm;
#pragma unroll
            for (int r = 0; r < 8; ++r) {
                int gm = m0 + wrow * 32 + mi * 16 + r + half * 8;
                float c = acc[mi][ni][r];
                if (bias_mode == 1)      c += bias[gn];
                else if (bias_mode == 2) c += bias[gm];
                if (act == 1) c = gelu_f(c);
                size_t idx = cb + (size_t)gm * ldc + gn;
                if (resid) c += resid[idx];
                if (outf) outf[idx] = c;
                if (outb) outb[idx] = f2bf(c);
            }
        }
}

// ---------------------------------------------------------------------------
// Spatial-gate pooling: pool over D -> [B,H,N,3] (max, mean, min)
// ---------------------------------------------------------------------------
__global__ void k_pool_sp(const float* __restrict__ qkv, float* __restrict__ pool) {
    int gid = blockIdx.x * blockDim.x + threadIdx.x;   // B*H*N
    int n = gid & 511, h = (gid >> 9) & 7, b = gid >> 12;
    const float* vp = qkv + ((size_t)(b * NN_ + n)) * (3 * EE) + 2 * EE + h * DD;
    float mx = -1e30f, mn = 1e30f, sm = 0.f;
    for (int d = 0; d < DD; ++d) {
        float v = vp[d];
        mx = fmaxf(mx, v); mn = fminf(mn, v); sm += v;
    }
    pool[(size_t)gid * 3 + 0] = mx;
    pool[(size_t)gid * 3 + 1] = sm * (1.0f / DD);
    pool[(size_t)gid * 3 + 2] = mn;
}

__global__ void k_sp_gate(const float* __restrict__ pool, const float* __restrict__ sp_w,
                          const float* __restrict__ sp_b, float* __restrict__ sa) {
    int gid = blockIdx.x * blockDim.x + threadIdx.x;   // B*H*N (o as head out)
    int n = gid & 511, o = (gid >> 9) & 7, b = gid >> 12;
    float t = 0.f;
#pragma unroll
    for (int j = 0; j < 3; ++j) {
        float a = sp_b[o];
#pragma unroll
        for (int h = 0; h < HH; ++h)
            a += sp_w[o * HH + h] * pool[(((size_t)(b * HH + h) * NN_) + n) * 3 + j];
        t += gelu_f(a);
    }
    sa[gid] = sigm_f(t);
}

// Spectral-gate pooling: pool over N -> [B,H,3,D]
__global__ void k_pool_sc(const float* __restrict__ qkv, float* __restrict__ sc) {
    int gid = blockIdx.x * blockDim.x + threadIdx.x;   // B*H*D
    int d = gid & 63, h = (gid >> 6) & 7, b = gid >> 9;
    float mx = -1e30f, mn = 1e30f, sm = 0.f;
    for (int n = 0; n < NN_; ++n) {
        float v = qkv[((size_t)(b * NN_ + n)) * (3 * EE) + 2 * EE + h * DD + d];
        mx = fmaxf(mx, v); mn = fminf(mn, v); sm += v;
    }
    size_t o = (size_t)(b * HH + h) * 3 * DD + d;
    sc[o]          = mx;
    sc[o + DD]     = sm * (1.0f / NN_);
    sc[o + 2 * DD] = mn;
}

__global__ void k_spec_gate(const float* __restrict__ sc, const float* __restrict__ w1,
                            const float* __restrict__ b1, const float* __restrict__ w2,
                            const float* __restrict__ b2, float* __restrict__ se) {
    int gid = blockIdx.x * blockDim.x + threadIdx.x;   // B*H
    if (gid >= BB * HH) return;
    const float* scp = sc + (size_t)gid * 3 * DD;
    float h1[3][32];
    for (int j = 0; j < 3; ++j)
        for (int p = 0; p < 32; ++p) {
            float a = b1[p];
            for (int d2 = 0; d2 < DD; ++d2) a += w1[p * DD + d2] * scp[j * DD + d2];
            h1[j][p] = gelu_f(a);
        }
    for (int d = 0; d < DD; ++d) {
        float t = 0.f;
        for (int j = 0; j < 3; ++j) {
            float a = b2[d];
            for (int p = 0; p < 32; ++p) a += w2[d * 32 + p] * h1[j][p];
            t += a;
        }
        se[(size_t)gid * DD + d] = sigm_f(t);
    }
}

// Repack qkv fp32 [B,N,3,H,D] -> bf16 q/k/v [B,H,N,D]; v scaled by gates;
// q pre-scaled by 1/sqrt(D) so the attention kernel skips per-tile scaling.
__global__ void k_build_qkv(const float* __restrict__ qkv, const float* __restrict__ sa,
                            const float* __restrict__ se, unsigned short* __restrict__ qb,
                            unsigned short* __restrict__ kb, unsigned short* __restrict__ vb) {
    int gid = blockIdx.x * blockDim.x + threadIdx.x;   // B*H*N*D
    int d = gid & 63, n = (gid >> 6) & 511, h = (gid >> 15) & 7, b = gid >> 18;
    size_t src = (size_t)(b * NN_ + n) * (3 * EE) + h * DD + d;
    float q = qkv[src];
    float k = qkv[src + EE];
    float v = qkv[src + 2 * EE];
    float g = sa[(size_t)(b * HH + h) * NN_ + n] * se[(size_t)(b * HH + h) * DD + d];
    qb[gid] = f2bf(q * 0.125f);
    kb[gid] = f2bf(k);
    vb[gid] = f2bf(v * g);
}

// ---------------------------------------------------------------------------
// Flash attention: per block one (b,h) x 64 q-rows. 4 waves, each 16 q-rows.
// S = Q K^T via WMMA (scale folded into Q), online softmax, O += P V via WMMA.
// ---------------------------------------------------------------------------
__global__ __launch_bounds__(128)
void k_attn(const unsigned short* __restrict__ qb, const unsigned short* __restrict__ kb,
            const unsigned short* __restrict__ vb, float* __restrict__ out) {
    int qt = blockIdx.x;               // 0..7
    int bh = blockIdx.y;               // 0..127
    int b = bh >> 3, h = bh & 7;
    size_t base = (size_t)bh * NN_ * DD;

    int wave = threadIdx.x >> 5;
    int lane = threadIdx.x & 31;
    int half = lane >> 4;
    int lm = lane & 15;

    __shared__ unsigned short sK[64 * 64];
    __shared__ unsigned short sVt[64 * 64];        // transposed: [d][kv]
    __shared__ unsigned short sP[4][16 * 64];      // per-wave P tile

    // Q fragments (2 K-chunks of 32), kept in registers
    U32u qf[2];
    {
        int qrow = qt * 64 + wave * 16 + lm;
        const unsigned short* qp = qb + base + (size_t)qrow * DD;
#pragma unroll
        for (int kc = 0; kc < 2; ++kc) {
            qf[kc].q[0] = *(const uint4*)(qp + kc * 32 + half * 8);
            qf[kc].q[1] = *(const uint4*)(qp + kc * 32 + half * 8 + 16);
        }
    }

    v8f o[4] = {};
    float m[8], lsum[8];
#pragma unroll
    for (int r = 0; r < 8; ++r) { m[r] = -1e30f; lsum[r] = 0.f; }

    for (int j = 0; j < 8; ++j) {
        __syncthreads();
        // ---- stage K (async) and V^T (transposed scatter) tiles ----
        {
            int row = threadIdx.x >> 1;
            int chunk = (threadIdx.x & 1) * 32;
            const unsigned short* kp = kb + base + (size_t)(j * 64 + row) * DD + chunk;
            unsigned short* skp = sK + row * 64 + chunk;
            cp16(kp,      skp);
            cp16(kp + 8,  skp + 8);
            cp16(kp + 16, skp + 16);
            cp16(kp + 24, skp + 24);
            U64u vt;
            const uint4* vp4 = (const uint4*)(vb + base + (size_t)(j * 64 + row) * DD + chunk);
            vt.q[0] = vp4[0]; vt.q[1] = vp4[1]; vt.q[2] = vp4[2]; vt.q[3] = vp4[3];
#pragma unroll
            for (int e = 0; e < 32; ++e) sVt[(chunk + e) * 64 + row] = vt.s[e];
        }
        async_wait();
        __syncthreads();

        // ---- S = Q K^T (scale pre-folded into Q) ----
        v8f s[4];
#pragma unroll
        for (int ni = 0; ni < 4; ++ni) {
            v8f z = {};
#pragma unroll
            for (int kc = 0; kc < 2; ++kc) {
                U32u bu;
                int n = ni * 16 + lm;
                bu.q[0] = *(const uint4*)(sK + n * 64 + kc * 32 + half * 8);
                bu.q[1] = *(const uint4*)(sK + n * 64 + kc * 32 + half * 8 + 16);
                z = wmma_bf16(qf[kc].v, bu.v, z);
            }
            s[ni] = z;
        }

        // ---- online softmax (row stats across 16 lanes of each half) ----
#pragma unroll
        for (int r = 0; r < 8; ++r) {
            float t = fmaxf(fmaxf(s[0][r], s[1][r]), fmaxf(s[2][r], s[3][r]));
            t = fmaxf(t, __shfl_xor(t, 1));
            t = fmaxf(t, __shfl_xor(t, 2));
            t = fmaxf(t, __shfl_xor(t, 4));
            t = fmaxf(t, __shfl_xor(t, 8));
            float mn = fmaxf(m[r], t);
            float sc = __expf(m[r] - mn);
            float rs = 0.f;
#pragma unroll
            for (int ni = 0; ni < 4; ++ni) {
                float e = __expf(s[ni][r] - mn);
                s[ni][r] = e;
                rs += e;
            }
            rs += __shfl_xor(rs, 1);
            rs += __shfl_xor(rs, 2);
            rs += __shfl_xor(rs, 4);
            rs += __shfl_xor(rs, 8);
            lsum[r] = lsum[r] * sc + rs;
#pragma unroll
            for (int ni = 0; ni < 4; ++ni) o[ni][r] *= sc;
            m[r] = mn;
        }

        // ---- P -> LDS (bf16, row-major 16x64), then O += P V ----
#pragma unroll
        for (int ni = 0; ni < 4; ++ni)
#pragma unroll
            for (int r = 0; r < 8; ++r)
                sP[wave][(r + half * 8) * 64 + ni * 16 + lm] = f2bf(s[ni][r]);
        __syncthreads();

#pragma unroll
        for (int ni = 0; ni < 4; ++ni)
#pragma unroll
            for (int kc = 0; kc < 2; ++kc) {
                U32u pa, vf;
                pa.q[0] = *(const uint4*)(sP[wave] + lm * 64 + kc * 32 + half * 8);
                pa.q[1] = *(const uint4*)(sP[wave] + lm * 64 + kc * 32 + half * 8 + 16);
                int d = ni * 16 + lm;
                vf.q[0] = *(const uint4*)(sVt + d * 64 + kc * 32 + half * 8);
                vf.q[1] = *(const uint4*)(sVt + d * 64 + kc * 32 + half * 8 + 16);
                o[ni] = wmma_bf16(pa.v, vf.v, o[ni]);
            }
    }

    // ---- normalize + write [B,N,E] ----
#pragma unroll
    for (int r = 0; r < 8; ++r) {
        float inv = 1.0f / (lsum[r] + 1e-20f);
        int gm = qt * 64 + wave * 16 + r + half * 8;
#pragma unroll
        for (int ni = 0; ni < 4; ++ni) {
            int d = ni * 16 + lm;
            out[((size_t)b * NN_ + gm) * EE + h * DD + d] = o[ni][r] * inv;
        }
    }
}

// ---------------------------------------------------------------------------
// Direct conv (k=3 / k=5) over channels N, length E; gelu -> bf16 into cat
// ---------------------------------------------------------------------------
__global__ void k_conv_gelu(const float* __restrict__ x, const float* __restrict__ w,
                            const float* __restrict__ bias, unsigned short* __restrict__ catbf,
                            int taps, int pad, int choff) {
    int gid = blockIdx.x * blockDim.x + threadIdx.x;    // B*C4*E
    int l = gid & 511, co = (gid >> 9) & 127, b = gid >> 16;
    float acc = bias[co];
    const float* xb = x + (size_t)b * NN_ * EE;
    const float* wr = w + (size_t)co * NN_ * taps;
    for (int ci = 0; ci < NN_; ++ci) {
        const float* xc = xb + (size_t)ci * EE;
        for (int t = 0; t < taps; ++t) {
            int ll = l + t - pad;
            if (ll >= 0 && ll < EE) acc = fmaf(wr[ci * taps + t], xc[ll], acc);
        }
    }
    catbf[((size_t)b * (4 * C4_) + choff + co) * EE + l] = f2bf(gelu_f(acc));
}

// maxpool3 along length -> bf16
__global__ void k_maxpool(const float* __restrict__ x, unsigned short* __restrict__ mpbf) {
    int gid = blockIdx.x * blockDim.x + threadIdx.x;    // B*N*E
    int l = gid & 511;
    float v = x[gid];
    if (l > 0)       v = fmaxf(v, x[gid - 1]);
    if (l < EE - 1)  v = fmaxf(v, x[gid + 1]);
    mpbf[gid] = f2bf(v);
}

__global__ void k_addy(float* __restrict__ x, const float* __restrict__ y) {
    int gid = blockIdx.x * blockDim.x + threadIdx.x;
    x[gid] += y[gid];
}

// ---------------------------------------------------------------------------
// Host orchestration
// ---------------------------------------------------------------------------
extern "C" void kernel_launch(void* const* d_in, const int* in_sizes, int n_in,
                              void* d_out, int out_size, void* d_ws, size_t ws_size,
                              hipStream_t stream) {
    const float* in_x     = (const float*)d_in[0];
    const float* in_y     = (const float*)d_in[1];
    const float* ln1_g    = (const float*)d_in[2];
    const float* ln1_b    = (const float*)d_in[3];
    const float* wqkv     = (const float*)d_in[4];
    const float* bqkv     = (const float*)d_in[5];
    const float* sp_w     = (const float*)d_in[6];
    const float* sp_b     = (const float*)d_in[7];
    const float* spec_w1  = (const float*)d_in[8];
    const float* spec_b1  = (const float*)d_in[9];
    const float* spec_w2  = (const float*)d_in[10];
    const float* spec_b2  = (const float*)d_in[11];
    const float* wout     = (const float*)d_in[12];
    const float* bout     = (const float*)d_in[13];
    const float* ln2_g    = (const float*)d_in[14];
    const float* ln2_b    = (const float*)d_in[15];
    const float* ff_w1    = (const float*)d_in[16];
    const float* ff_b1    = (const float*)d_in[17];
    const float* ff_w2    = (const float*)d_in[18];
    const float* ff_b2    = (const float*)d_in[19];
    const float* inc1_w   = (const float*)d_in[20];
    const float* inc1_b   = (const float*)d_in[21];
    const float* inc2_w   = (const float*)d_in[22];
    const float* inc2_b   = (const float*)d_in[23];
    const float* inc3_w   = (const float*)d_in[24];
    const float* inc3_b   = (const float*)d_in[25];
    const float* inc4_w   = (const float*)d_in[26];
    const float* inc4_b   = (const float*)d_in[27];
    const float* incp_w   = (const float*)d_in[28];
    const float* incp_b   = (const float*)d_in[29];
    (void)in_sizes; (void)n_in; (void)out_size; (void)ws_size;

    char* wsp = (char*)d_ws;
    size_t off = 0;
    auto alloc = [&](size_t bytes) -> void* {
        void* p = wsp + off;
        off += (bytes + 255) & ~(size_t)255;
        return p;
    };

    const size_t TOK = (size_t)BB * NN_;       // 8192 rows
    const size_t XEL = TOK * EE;               // 4,194,304

    unsigned short* wqkv_bf = (unsigned short*)alloc((size_t)DEPTH_ * 3 * EE * EE * 2);
    unsigned short* wout_bf = (unsigned short*)alloc((size_t)DEPTH_ * EE * EE * 2);
    unsigned short* ffw1_bf = (unsigned short*)alloc((size_t)DEPTH_ * MLP_ * EE * 2);
    unsigned short* ffw2_bf = (unsigned short*)alloc((size_t)DEPTH_ * EE * MLP_ * 2);
    unsigned short* inc1_bf = (unsigned short*)alloc((size_t)C4_ * NN_ * 2);
    unsigned short* inc4_bf = (unsigned short*)alloc((size_t)C4_ * NN_ * 2);
    unsigned short* incp_bf = (unsigned short*)alloc((size_t)NN_ * (4 * C4_) * 2);

    float*          xbuf    = (float*)alloc(XEL * 4);
    unsigned short* xn_bf   = (unsigned short*)alloc(XEL * 2);
    float*          qkvb    = (float*)alloc(TOK * 3 * EE * 4);
    float*          poolsp  = (float*)alloc((size_t)BB * HH * NN_ * 3 * 4);
    float*          sabuf   = (float*)alloc((size_t)BB * HH * NN_ * 4);
    float*          scbuf   = (float*)alloc((size_t)BB * HH * 3 * DD * 4);
    float*          sebuf   = (float*)alloc((size_t)BB * HH * DD * 4);
    unsigned short* qbf     = (unsigned short*)alloc(XEL * 2);
    unsigned short* kbf     = (unsigned short*)alloc(XEL * 2);
    unsigned short* vbf     = (unsigned short*)alloc(XEL * 2);
    float*          attno   = (float*)alloc(XEL * 4);
    unsigned short* attnobf = (unsigned short*)alloc(XEL * 2);
    unsigned short* xbf     = (unsigned short*)alloc(XEL * 2);
    unsigned short* catbf   = (unsigned short*)alloc(XEL * 2);
    unsigned short* mpbf    = (unsigned short*)alloc(XEL * 2);
    unsigned short* hn_bf   = (unsigned short*)alloc(XEL * 2);
    unsigned short* hbuf_bf = (unsigned short*)alloc(TOK * MLP_ * 2);

    auto cvt = [&](const float* s, unsigned short* d, size_t n) {
        k_cvt<<<dim3((unsigned)((n + 255) / 256)), dim3(256), 0, stream>>>(s, d, (int)n);
    };

    // weights -> bf16 (all layers at once)
    cvt(wqkv,   wqkv_bf, (size_t)DEPTH_ * 3 * EE * EE);
    cvt(wout,   wout_bf, (size_t)DEPTH_ * EE * EE);
    cvt(ff_w1,  ffw1_bf, (size_t)DEPTH_ * MLP_ * EE);
    cvt(ff_w2,  ffw2_bf, (size_t)DEPTH_ * EE * MLP_);
    cvt(inc1_w, inc1_bf, (size_t)C4_ * NN_);
    cvt(inc4_w, inc4_bf, (size_t)C4_ * NN_);
    cvt(incp_w, incp_bf, (size_t)NN_ * (4 * C4_));

    // x working copy (must not mutate inputs)
    (void)hipMemcpyAsync(xbuf, in_x, XEL * 4, hipMemcpyDeviceToDevice, stream);

    auto gemm = [&](const unsigned short* A, int lda, long long bA,
                    const unsigned short* Bm, int ldb, long long bB,
                    const float* bias, int bias_mode, const float* resid,
                    float* outf, unsigned short* outb, int ldc, long long bC,
                    int M, int Nn, int K, int bmode, int act, int batch) {
        dim3 g((unsigned)(Nn / 128), (unsigned)(M / 64), (unsigned)batch);
        k_gemm<<<g, dim3(256), 0, stream>>>(A, lda, bA, Bm, ldb, bB, bias, bias_mode,
                                            resid, outf, outb, ldc, bC, M, Nn, K, bmode, act);
    };

    const long long XB = (long long)NN_ * EE;   // per-batch plane stride

    for (int i = 0; i < DEPTH_; ++i) {
        // ---- PreNorm + MHA ----
        k_layernorm<<<dim3((unsigned)TOK), dim3(256), 0, stream>>>(
            xbuf, ln1_g + i * EE, ln1_b + i * EE, xn_bf);

        gemm(xn_bf, EE, 0, wqkv_bf + (size_t)i * 3 * EE * EE, EE, 0,
             bqkv + i * 3 * EE, 1, nullptr, qkvb, nullptr, 3 * EE, 0,
             (int)TOK, 3 * EE, EE, /*NT*/0, /*none*/0, 1);

        k_pool_sp<<<dim3(BB * HH * NN_ / 256), dim3(256), 0, stream>>>(qkvb, poolsp);
        k_sp_gate<<<dim3(BB * HH * NN_ / 256), dim3(256), 0, stream>>>(
            poolsp, sp_w + i * HH * HH, sp_b + i * HH, sabuf);
        k_pool_sc<<<dim3(BB * HH * DD / 256), dim3(256), 0, stream>>>(qkvb, scbuf);
        k_spec_gate<<<dim3(1), dim3(128), 0, stream>>>(
            scbuf, spec_w1 + i * (DD / 2) * DD, spec_b1 + i * (DD / 2),
            spec_w2 + i * DD * (DD / 2), spec_b2 + i * DD, sebuf);
        k_build_qkv<<<dim3((unsigned)(XEL / 256)), dim3(256), 0, stream>>>(
            qkvb, sabuf, sebuf, qbf, kbf, vbf);

        k_attn<<<dim3(NN_ / 64, BB * HH), dim3(128), 0, stream>>>(qbf, kbf, vbf, attno);
        cvt(attno, attnobf, XEL);

        gemm(attnobf, EE, 0, wout_bf + (size_t)i * EE * EE, EE, 0,
             bout + i * EE, 1, /*resid*/xbuf, xbuf, nullptr, EE, 0,
             (int)TOK, EE, EE, 0, 0, 1);

        // ---- Inception (shared weights, no residual) ----
        cvt(xbuf, xbf, XEL);
        gemm(inc1_bf, NN_, 0, xbf, EE, XB, inc1_b, 2, nullptr,
             nullptr, catbf + 0 * C4_ * EE, EE, (long long)(4 * C4_) * EE,
             C4_, EE, NN_, /*NN*/1, /*gelu*/1, BB);
        k_conv_gelu<<<dim3((unsigned)(BB * C4_ * EE / 256)), dim3(256), 0, stream>>>(
            xbuf, inc2_w, inc2_b, catbf, 3, 1, 1 * C4_);
        k_conv_gelu<<<dim3((unsigned)(BB * C4_ * EE / 256)), dim3(256), 0, stream>>>(
            xbuf, inc3_w, inc3_b, catbf, 5, 2, 2 * C4_);
        k_maxpool<<<dim3((unsigned)(XEL / 256)), dim3(256), 0, stream>>>(xbuf, mpbf);
        gemm(inc4_bf, NN_, 0, mpbf, EE, XB, inc4_b, 2, nullptr,
             nullptr, catbf + 3 * C4_ * EE, EE, (long long)(4 * C4_) * EE,
             C4_, EE, NN_, 1, 1, BB);
        gemm(incp_bf, 4 * C4_, 0, catbf, EE, (long long)(4 * C4_) * EE, incp_b, 2, nullptr,
             xbuf, nullptr, EE, XB,
             NN_, EE, 4 * C4_, 1, 1, BB);

        // ---- PreNorm + FFN ----
        k_layernorm<<<dim3((unsigned)TOK), dim3(256), 0, stream>>>(
            xbuf, ln2_g + i * EE, ln2_b + i * EE, hn_bf);
        gemm(hn_bf, EE, 0, ffw1_bf + (size_t)i * MLP_ * EE, EE, 0,
             ff_b1 + i * MLP_, 1, nullptr, nullptr, hbuf_bf, MLP_, 0,
             (int)TOK, MLP_, EE, 0, /*gelu*/1, 1);
        gemm(hbuf_bf, MLP_, 0, ffw2_bf + (size_t)i * EE * MLP_, MLP_, 0,
             ff_b2 + i * EE, 1, /*resid*/xbuf, xbuf, nullptr, EE, 0,
             (int)TOK, EE, MLP_, 0, 0, 1);

        if (i < DEPTH_ - 1)
            k_addy<<<dim3((unsigned)(XEL / 256)), dim3(256), 0, stream>>>(xbuf, in_y);
    }

    (void)hipMemcpyAsync(d_out, xbuf, XEL * 4, hipMemcpyDeviceToDevice, stream);
}
